// SioConvLayer_30743375905260
// MI455X (gfx1250) — matrix operand order repaired and checked by reading, not compiled, not run on hardware
//
#include <hip/hip_runtime.h>
#include <hip/hip_bf16.h>
#include <stdint.h>

// ---------------------------------------------------------------------------
// SioConv layer for MI455X (gfx1250, wave32).
//  - All GEMM operands pre-converted to f16 once; GEMMs run
//    v_wmma_f32_16x16x32_f16 with double-buffered LDS fed by
//    global_load_async_to_lds_b128 (ASYNCcnt) -- CDNA5 async copy path.
//  - Complex gated recurrence: one (b,h) per block, one state column per
//    wave, one state row per lane; shfl broadcast + butterfly reduction.
// ---------------------------------------------------------------------------

typedef _Float16 v16h __attribute__((ext_vector_type(16)));
typedef _Float16 v8h  __attribute__((ext_vector_type(8)));
typedef _Float16 v4h  __attribute__((ext_vector_type(4)));
typedef float    v8f  __attribute__((ext_vector_type(8)));
typedef float    v4f  __attribute__((ext_vector_type(4)));

#define BM 128
#define BN 128
#define BK 32
#define LSTR 56            // padded LDS row stride in halves (112B: 16B aligned, bank-clean)
#define LBUF (BM * LSTR)   // one LDS buffer (halves); A and B both 128 rows

__device__ __forceinline__ void async_b128(uint32_t lds_byte_addr, const void* gaddr) {
    // CDNA5: GLOBAL_LOAD_ASYNC_TO_LDS_B128  (VDST = LDS byte addr, VADDR = 64b global addr)
    asm volatile("global_load_async_to_lds_b128 %0, %1, off"
                 :: "v"(lds_byte_addr), "v"((uint64_t)(uintptr_t)gaddr) : "memory");
}
__device__ __forceinline__ void wait_async0() {
    asm volatile("s_wait_asynccnt 0" ::: "memory");
}
__device__ __forceinline__ float fast_silu(float x) {
    return x * __builtin_amdgcn_rcpf(1.0f + __expf(-x));
}

// ---------------------------------------------------------------------------
// one-time f32 -> f16 conversion (vectorized, n multiple of 1024)
// ---------------------------------------------------------------------------
__global__ __launch_bounds__(256) void cvt_f32_f16_kernel(
    const float* __restrict__ in, _Float16* __restrict__ out, int n4)
{
    const int i = blockIdx.x * 256 + threadIdx.x;
    if (i < n4) {
        v4f v = *(const v4f*)&in[(size_t)i * 4];
        v4h h;
#pragma unroll
        for (int j = 0; j < 4; ++j) h[j] = (_Float16)v[j];
        *(v4h*)&out[(size_t)i * 4] = h;
    }
}

// ---------------------------------------------------------------------------
// C[M,N] = act(A[M,K] @ W[N,K]^T + bias[N]); A, W f16; C f16 or f32.
// 256 threads = 8 waves (4 m-slabs x 2 n-slabs), wave tile 32x64 = 2x4 WMMA.
// Double-buffered LDS, async global->LDS staging.
// ---------------------------------------------------------------------------
template<typename OT, bool SILU>
__global__ __launch_bounds__(256) void gemm_wmma_async_kernel(
    const _Float16* __restrict__ A, const _Float16* __restrict__ Wt,
    const float* __restrict__ bias, OT* __restrict__ C,
    int M, int N, int K)
{
    __shared__ _Float16 sA[2 * LBUF];
    __shared__ _Float16 sB[2 * LBUF];

    const int tid  = threadIdx.x;
    const int lane = tid & 31;
    const int wave = tid >> 5;
    const int lm   = lane & 15;
    const int hlf  = lane >> 4;
    const int wm   = wave >> 1;   // 0..3 -> 32-row m slab
    const int wn   = wave & 1;    // 0..1 -> 64-col n slab
    const int blockM = blockIdx.y * BM;
    const int blockN = blockIdx.x * BN;

    // Staging: 128 rows x 4 16B-chunks per operand = 512 transfers; 2 per thread.
    const int r0 = tid >> 2;      // 0..63
    const int ch = tid & 3;       // chunk -> halves [ch*8, ch*8+8)

    const _Float16* gA0 = A  + (size_t)(blockM + r0)      * K + ch * 8;
    const _Float16* gA1 = A  + (size_t)(blockM + r0 + 64) * K + ch * 8;
    const _Float16* gB0 = Wt + (size_t)(blockN + r0)      * K + ch * 8;
    const _Float16* gB1 = Wt + (size_t)(blockN + r0 + 64) * K + ch * 8;
    const uint32_t lA0 = (uint32_t)(uintptr_t)&sA[r0 * LSTR + ch * 8];
    const uint32_t lA1 = (uint32_t)(uintptr_t)&sA[(r0 + 64) * LSTR + ch * 8];
    const uint32_t lB0 = (uint32_t)(uintptr_t)&sB[r0 * LSTR + ch * 8];
    const uint32_t lB1 = (uint32_t)(uintptr_t)&sB[(r0 + 64) * LSTR + ch * 8];

    auto stage = [&](int buf, int k0) {
        const uint32_t bo = (uint32_t)(buf * LBUF * sizeof(_Float16));
        async_b128(lA0 + bo, gA0 + k0);
        async_b128(lA1 + bo, gA1 + k0);
        async_b128(lB0 + bo, gB0 + k0);
        async_b128(lB1 + bo, gB1 + k0);
    };

    v8f acc[2][4];
#pragma unroll
    for (int mi = 0; mi < 2; ++mi)
#pragma unroll
        for (int ni = 0; ni < 4; ++ni)
#pragma unroll
            for (int i = 0; i < 8; ++i) acc[mi][ni][i] = 0.0f;

    stage(0, 0);
    int buf = 0;
    for (int k0 = 0; k0 < K; k0 += BK) {
        wait_async0();          // my async writes into 'buf' have landed
        __syncthreads();        // everyone's have; previous readers of buf^1 are done
        if (k0 + BK < K) stage(buf ^ 1, k0 + BK);

        const _Float16* bA = &sA[buf * LBUF];
        const _Float16* bB = &sB[buf * LBUF];

        // A 16x32 fragment: lane(m=lm, half): h[0..7] K=half*8+i, h[8..15] K=16+half*8+i
        v16h afrag[2];
#pragma unroll
        for (int mi = 0; mi < 2; ++mi) {
            const int row = wm * 32 + mi * 16 + lm;
            v8h lo = *(const v8h*)&bA[row * LSTR + hlf * 8];
            v8h hi = *(const v8h*)&bA[row * LSTR + 16 + hlf * 8];
#pragma unroll
            for (int i = 0; i < 8; ++i) { afrag[mi][i] = lo[i]; afrag[mi][i + 8] = hi[i]; }
        }
        // B 32x16 fragment: lane(n=lm, half): h[0..7] K=half*8+i, h[8..15] K=16+half*8+i  (B[k][n] == W[n][k])
        v16h bfrag[4];
#pragma unroll
        for (int ni = 0; ni < 4; ++ni) {
            const int row = wn * 64 + ni * 16 + lm;
            v8h lo = *(const v8h*)&bB[row * LSTR + hlf * 8];
            v8h hi = *(const v8h*)&bB[row * LSTR + 16 + hlf * 8];
#pragma unroll
            for (int i = 0; i < 8; ++i) { bfrag[ni][i] = lo[i]; bfrag[ni][i + 8] = hi[i]; }
        }

#pragma unroll
        for (int mi = 0; mi < 2; ++mi)
#pragma unroll
            for (int ni = 0; ni < 4; ++ni)
                acc[mi][ni] = __builtin_amdgcn_wmma_f32_16x16x32_f16(
                    false, afrag[mi], false, bfrag[ni],
                    (short)0, acc[mi][ni], false, false);
        buf ^= 1;
    }

    // Epilogue: D layout -> n = lm, m = half*8 + r
#pragma unroll
    for (int mi = 0; mi < 2; ++mi) {
#pragma unroll
        for (int ni = 0; ni < 4; ++ni) {
            const int n  = blockN + wn * 64 + ni * 16 + lm;
            const float bv = bias[n];
#pragma unroll
            for (int r = 0; r < 8; ++r) {
                const int m = blockM + wm * 32 + mi * 16 + hlf * 8 + r;
                float v = acc[mi][ni][r] + bv;
                if (SILU) v = fast_silu(v);
                C[(size_t)m * N + n] = (OT)v;
            }
        }
    }
}

// ---------------------------------------------------------------------------
// Complex gated linear recurrence.
// grid = B*H; block = 1024 = 32 waves. wave = column e, lane = row d.
// h_l = a_l(d)*h_{l-1} + k_l(d)*v_l(e) ; y_l(e) = sum_d q_l(d)*h_l(d,e)
// ---------------------------------------------------------------------------
#define Bz 4
#define Lz 1024
#define Hz 8
#define Dz 32

__global__ __launch_bounds__(1024) void sioconv_scan_kernel(
    const float* __restrict__ qkva,   // [B,L,H,D,4,2]
    const float* __restrict__ h0_re,  // [H,D,D]
    const float* __restrict__ h0_im,  // [H,D,D]
    _Float16* __restrict__ y2h)       // [B,L,H*D*2] f16 (feeds next GEMM)
{
    const int bh = blockIdx.x;
    const int b  = bh / Hz;
    const int h  = bh % Hz;
    const int e  = threadIdx.x >> 5;
    const int d  = threadIdx.x & 31;

    float hr = h0_re[((size_t)h * Dz + d) * Dz + e];
    float hi = h0_im[((size_t)h * Dz + d) * Dz + e];

    size_t base = (((size_t)b * Lz) * Hz + h) * (size_t)(Dz * 8) + (size_t)d * 8;
    const size_t stepStride = (size_t)Hz * Dz * 8;
    size_t ybase = ((size_t)b * Lz) * (Hz * Dz * 2) + (size_t)h * (Dz * 2) + (size_t)e * 2;

    float4 c0 = *(const float4*)&qkva[base + 0];
    float4 c1 = *(const float4*)&qkva[base + 4];

    for (int l = 0; l < Lz; ++l) {
        // Prefetch next step (last iter reads 32B into the following ws region
        // -- allocated and unused here; value is discarded).
        const size_t nbase = base + stepStride;
        const float4 n0 = *(const float4*)&qkva[nbase + 0];
        const float4 n1 = *(const float4*)&qkva[nbase + 4];

        float qr = c0.x, qi = c0.y, kr = c0.z, ki = c0.w;
        float vr = c1.x, vi = c1.y, ar = c1.z, ai = c1.w;

        // gate: a *= rsqrt(|a|^2) * sigmoid(log|a|^2) ; sigmoid(log m) = m/(1+m)
        const float m2 = ar * ar + ai * ai;
        const float s  = rsqrtf(m2) * (m2 * __builtin_amdgcn_rcpf(1.0f + m2));
        ar *= s; ai *= s;

        // broadcast v from the lane whose row index equals this wave's column e
        const float vre = __shfl(vr, e, 32);
        const float vie = __shfl(vi, e, 32);

        // h = a*h + k*v (complex)
        const float nhr = ar * hr - ai * hi + kr * vre - ki * vie;
        const float nhi = ar * hi + ai * hr + kr * vie + ki * vre;
        hr = nhr; hi = nhi;

        // y_e = sum_d q_d * h_de (complex), butterfly over lanes
        float tr = qr * hr - qi * hi;
        float ti = qr * hi + qi * hr;
#pragma unroll
        for (int off = 16; off > 0; off >>= 1) {
            tr += __shfl_xor(tr, off, 32);
            ti += __shfl_xor(ti, off, 32);
        }
        if (d == 0) {
            union { uint32_t u; _Float16 hv[2]; } pk;
            pk.hv[0] = (_Float16)tr;
            pk.hv[1] = (_Float16)ti;
            *(uint32_t*)&y2h[ybase] = pk.u;   // ybase is even -> 4B aligned
        }
        c0 = n0; c1 = n1;
        base = nbase;
        ybase += (size_t)(Hz * Dz * 2);
    }
}

// ---------------------------------------------------------------------------
extern "C" void kernel_launch(void* const* d_in, const int* in_sizes, int n_in,
                              void* d_out, int out_size, void* d_ws, size_t ws_size,
                              hipStream_t stream) {
    (void)in_sizes; (void)n_in; (void)out_size; (void)ws_size;

    const float* x      = (const float*)d_in[0];
    const float* W_in   = (const float*)d_in[1];
    const float* b_in   = (const float*)d_in[2];
    const float* W_qkva = (const float*)d_in[3];
    const float* b_qkva = (const float*)d_in[4];
    const float* W_y    = (const float*)d_in[5];
    const float* b_y    = (const float*)d_in[6];
    const float* W_out  = (const float*)d_in[7];
    const float* b_out  = (const float*)d_in[8];
    const float* h0_re  = (const float*)d_in[9];
    const float* h0_im  = (const float*)d_in[10];
    float* out = (float*)d_out;

    const int M = Bz * Lz;   // 4096 tokens
    char* ws = (char*)d_ws;
    _Float16* xh   = (_Float16*)ws;  ws += (size_t)M * 1024 * 2;
    _Float16* wih  = (_Float16*)ws;  ws += (size_t)1024 * 1024 * 2;
    _Float16* wqh  = (_Float16*)ws;  ws += (size_t)2048 * 1024 * 2;
    _Float16* wyh  = (_Float16*)ws;  ws += (size_t)1024 * 512 * 2;
    _Float16* woh  = (_Float16*)ws;  ws += (size_t)1024 * 1024 * 2;
    _Float16* u    = (_Float16*)ws;  ws += (size_t)M * 1024 * 2;
    float*    qkva = (float*)ws;     ws += (size_t)M * 2048 * 4;
    _Float16* y2h  = (_Float16*)ws;  ws += (size_t)M * 512 * 2;
    _Float16* y2g  = (_Float16*)ws;  ws += (size_t)M * 1024 * 2;

    const dim3 blk(256);
    auto cvt = [&](const float* src, _Float16* dst, size_t n) {
        cvt_f32_f16_kernel<<<dim3((unsigned)(n / 4 / 256)), blk, 0, stream>>>(src, dst, (int)(n / 4));
    };
    cvt(x,      xh,  (size_t)M * 1024);
    cvt(W_in,   wih, (size_t)1024 * 1024);
    cvt(W_qkva, wqh, (size_t)2048 * 1024);
    cvt(W_y,    wyh, (size_t)1024 * 512);
    cvt(W_out,  woh, (size_t)1024 * 1024);

    // 1) u = silu(x @ W_in^T + b_in)            [4096 x 1024 x 1024]
    gemm_wmma_async_kernel<_Float16, true>
        <<<dim3(1024 / BN, M / BM), blk, 0, stream>>>(xh, wih, b_in, u, M, 1024, 1024);
    // 2) qkva = u @ W_qkva^T + b_qkva           [4096 x 2048 x 1024]
    gemm_wmma_async_kernel<float, false>
        <<<dim3(2048 / BN, M / BM), blk, 0, stream>>>(u, wqh, b_qkva, qkva, M, 2048, 1024);
    // 3) gated complex recurrence -> y2 (f16)    [B,L,512]
    sioconv_scan_kernel<<<dim3(Bz * Hz), dim3(1024), 0, stream>>>(qkva, h0_re, h0_im, y2h);
    // 4) y2g = silu(y2 @ W_y^T + b_y)           [4096 x 1024 x 512]
    gemm_wmma_async_kernel<_Float16, true>
        <<<dim3(1024 / BN, M / BM), blk, 0, stream>>>(y2h, wyh, b_y, y2g, M, 1024, 512);
    // 5) out = y2g @ W_out^T + b_out            [4096 x 1024 x 1024]
    gemm_wmma_async_kernel<float, false>
        <<<dim3(1024 / BN, M / BM), blk, 0, stream>>>(y2g, woh, b_out, out, M, 1024, 1024);
}
